// CSC1d_12438225289761
// MI455X (gfx1250) — matrix operand order repaired and checked by reading, not compiled, not run on hardware
//
#include <hip/hip_runtime.h>
#include <math.h>

// ---------------- problem constants ----------------
#define NB    8        // batch
#define CH    16       // channels
#define T_LEN 32768    // time
#define KA    32       // atoms
#define KS    64       // kernel size
#define TZ    32705    // T - KS + 1
#define TZP   32832    // 63 + TZ + 64  (zero-padded z length, mult of 64)
#define QC    1024     // conv reduction  = CH*KS
#define QT    2048     // convt reduction = KA*KS
#define LMBD  0.1f

typedef _Float16 half8 __attribute__((ext_vector_type(8)));
typedef _Float16 v16h  __attribute__((ext_vector_type(16)));
typedef float    v8f   __attribute__((ext_vector_type(8)));
typedef int      v4i   __attribute__((ext_vector_type(4)));

// ---- CDNA5 async memory->LDS path (guarded; falls back to sync copy) ----
#if defined(__has_builtin)
#if __has_builtin(__builtin_amdgcn_global_load_async_to_lds_b128) && \
    __has_builtin(__builtin_amdgcn_s_wait_asynccnt)
#define USE_ASYNC_LDS 1
#endif
#endif
#ifndef USE_ASYNC_LDS
#define USE_ASYNC_LDS 0
#endif

__device__ __forceinline__ void cp16_g2lds(const _Float16* g, _Float16* l) {
#if USE_ASYNC_LDS
    __builtin_amdgcn_global_load_async_to_lds_b128(
        (__attribute__((address_space(1))) v4i*)g,
        (__attribute__((address_space(3))) v4i*)l, 0, 0);
#else
    *(uint4*)l = *(const uint4*)g;
#endif
}

__device__ __forceinline__ void wait_async_lds() {
#if USE_ASYNC_LDS
    __builtin_amdgcn_s_wait_asynccnt(0);
#endif
}

// ---------------- utility kernels ----------------
__global__ void k_zero(unsigned int* __restrict__ p, unsigned int nwords) {
    unsigned int i = blockIdx.x * blockDim.x + threadIdx.x;
    unsigned int stride = gridDim.x * blockDim.x;
    for (; i < nwords; i += stride) p[i] = 0u;
}

// Build f16 GEMM "A" matrices from D (f32, layout [ko][c][j]):
//   Dconv [ko][c*64+j]  = D[ko][c][j]          (same flat order)
//   DconvT[c][ko*64+jj] = D[ko][c][63-jj]      (flipped for transpose-conv)
__global__ void k_prepD(const float* __restrict__ D,
                        _Float16* __restrict__ Dconv,
                        _Float16* __restrict__ DconvT) {
    int idx = blockIdx.x * blockDim.x + threadIdx.x;   // 32768 total
    if (idx < KA * CH * KS) {
        float v = D[idx];
        Dconv[idx] = (_Float16)v;
        int j  = idx & 63;
        int c  = (idx >> 6) & 15;
        int ko = idx >> 10;
        DconvT[(size_t)c * QT + ko * KS + (63 - j)] = (_Float16)v;
    }
}

// L = sum over (atom,channel) rows of max_f |DFT64(row)|^2 ; L==0 -> 1
__global__ void k_lipschitz(const float* __restrict__ D, float* __restrict__ Lout) {
    __shared__ float red[256];
    float s = 0.0f;
    for (int r = threadIdx.x; r < KA * CH; r += 256) {
        const float* d = D + (size_t)r * KS;
        float dv[KS];
        #pragma unroll
        for (int j = 0; j < KS; ++j) dv[j] = d[j];
        float mx = 0.0f;
        for (int f = 0; f < KS; ++f) {
            float re = 0.0f, im = 0.0f;
            for (int j = 0; j < KS; ++j) {
                float ang = -6.283185307179586f * (float)(f * j) / 64.0f;
                re += dv[j] * __cosf(ang);
                im += dv[j] * __sinf(ang);
            }
            mx = fmaxf(mx, re * re + im * im);
        }
        s += mx;
    }
    red[threadIdx.x] = s;
    __syncthreads();
    for (int off = 128; off > 0; off >>= 1) {
        if (threadIdx.x < off) red[threadIdx.x] += red[threadIdx.x + off];
        __syncthreads();
    }
    if (threadIdx.x == 0) {
        float L = red[0];
        Lout[0] = (L == 0.0f) ? 1.0f : L;
    }
}

// ---------------- LDS staging (async where possible) ----------------
// Stage NROWS x RS halves of signal (row stride rsG halves) starting at col t0.
// Out-of-range columns -> 0.
template<int NROWS, int RS>
__device__ __forceinline__ void stage_sig(_Float16* __restrict__ sSig,
                                          const _Float16* __restrict__ gbase,
                                          int rsG, int t0, int sigLen) {
    const int chunksPerRow = RS / 8;
    for (int idx = threadIdx.x; idx < NROWS * chunksPerRow; idx += 256) {
        int row = idx / chunksPerRow;
        int ch  = idx % chunksPerRow;
        int col = t0 + ch * 8;
        if (col + 8 <= sigLen) {
            cp16_g2lds(gbase + (size_t)row * rsG + col, &sSig[row * RS + ch * 8]);
        } else {
            #pragma unroll
            for (int e = 0; e < 8; ++e)
                sSig[row * RS + ch * 8 + e] =
                    (col + e < sigLen) ? gbase[(size_t)row * rsG + col + e]
                                       : (_Float16)0.0f;
        }
    }
    wait_async_lds();
}

// 16-bit A fragment (16x32): lanes 0-15 K 0..15,
// lanes 16-31 K 16..31; built from row-major f16 matrix (row stride Q).
template<int Q>
__device__ __forceinline__ v16h load_a_frag(const _Float16* __restrict__ Ag,
                                            int m, int q0, int hf) {
    const half8* A8 = (const half8*)Ag;
    size_t base = ((size_t)m * Q + q0) >> 3;       // half8 units; 16B aligned
    half8 lo = A8[base + 2 * hf];
    half8 hi = A8[base + 2 * hf + 1];
    return __builtin_shufflevector(lo, hi, 0,1,2,3,4,5,6,7,8,9,10,11,12,13,14,15);
}

// B fragment (32x16): lane n = column t; lanes 0-15 hold K 0..15, 16-31 K 16..31
// -> 16 contiguous halves of the sliding window in LDS.
__device__ __forceinline__ v16h load_b_frag(const _Float16* __restrict__ bp) {
    v16h b;
    #pragma unroll
    for (int i = 0; i < 16; ++i) b[i] = bp[i];
    return b;
}

// ---------------- convt kernel: out[c,t] = sum_{ko,jj} zpad[ko][t+jj]*DT[c][ko*64+jj]
// MODE 0: r = convt(z) - x  (store f16)      MODE 1: out = convt(w)  (store f32)
// Block: 256 time points; each wave owns 2 time subtiles sharing one A fragment.
template<int MODE>
__global__ void __launch_bounds__(256)
k_convt(const _Float16* __restrict__ sig,     // zpad or wpad, [NB][KA][TZP]
        const _Float16* __restrict__ Ag,      // DconvT [CH][QT]
        const float* __restrict__ x,
        _Float16* __restrict__ r_out,
        float* __restrict__ f32_out) {
    constexpr int TB = 256;
    constexpr int RS = TB + KS;               // 320, mult of 8
    __shared__ __align__(16) _Float16 sSig[KA * RS];   // 20 KB

    const int t0 = blockIdx.x * TB;
    const int n  = blockIdx.y;
    const _Float16* gbase = sig + (size_t)n * KA * TZP;

    stage_sig<KA, RS>(sSig, gbase, TZP, t0, TZP);
    __syncthreads();

    const int lane = threadIdx.x & 31;
    const int wave = threadIdx.x >> 5;
    const int nl = lane & 15, hf = lane >> 4;
    const int tt0 = wave * 32;                // two 16-wide subtiles per wave

    v8f acc0 = {}, acc1 = {};
    #pragma unroll 4
    for (int kk = 0; kk < QT / 32; ++kk) {            // 64 steps, 128 WMMAs/wave
        int q0 = kk * 32;
        int row = kk >> 1;                            // atom index
        int jbase = (kk & 1) * 32;
        v16h a  = load_a_frag<QT>(Ag, nl, q0, hf);
        const _Float16* bp = &sSig[row * RS + tt0 + nl + jbase + hf * 16];
        v16h b0 = load_b_frag(bp);
        v16h b1 = load_b_frag(bp + 16);
        acc0 = __builtin_amdgcn_wmma_f32_16x16x32_f16(
                   false, a, false, b0, (short)0, acc0, false, false);
        acc1 = __builtin_amdgcn_wmma_f32_16x16x32_f16(
                   false, a, false, b1, (short)0, acc1, false, false);
    }

    const int ta = t0 + tt0 + nl;
    #pragma unroll
    for (int i = 0; i < 8; ++i) {
        int c = i + 8 * hf;                            // C/D layout: M = i + 8*half
        size_t base = ((size_t)(n * CH + c)) * T_LEN;
        if (MODE == 0) {
            r_out[base + ta]      = (_Float16)(acc0[i] - x[base + ta]);
            r_out[base + ta + 16] = (_Float16)(acc1[i] - x[base + ta + 16]);
        } else {
            f32_out[base + ta]      = acc0[i];
            f32_out[base + ta + 16] = acc1[i];
        }
    }
}

// ---------------- conv + prox kernel ----------------
// grad[ko,t] = sum_{c,j} r[c][t+j]*Dconv[ko][c*64+j];
// w_new = relu(z - grad/L - lmbd/L); z_new = w_new + mom*(w_new - w_old)
// Block: 128 time points; wave = (ko half) x (2 time subtiles sharing A).
__global__ void __launch_bounds__(256)
k_conv_prox(const _Float16* __restrict__ r,      // [NB][CH][T_LEN] f16
            const _Float16* __restrict__ Ag,     // Dconv [KA][QC]
            const float* __restrict__ Lptr,
            float* __restrict__ z32, float* __restrict__ w32,
            _Float16* __restrict__ zpad, _Float16* __restrict__ wpad,
            float mom) {
    constexpr int TB = 128;
    constexpr int RS = TB + KS;                  // 192
    __shared__ __align__(16) _Float16 sSig[CH * RS];   // 6 KB

    const int t0 = blockIdx.x * TB;
    const int n  = blockIdx.y;
    const _Float16* gbase = r + (size_t)n * CH * T_LEN;

    stage_sig<CH, RS>(sSig, gbase, T_LEN, t0, T_LEN);
    __syncthreads();

    const int lane  = threadIdx.x & 31;
    const int wave  = threadIdx.x >> 5;
    const int nl = lane & 15, hf = lane >> 4;
    const int mtile = wave & 1;                  // ko half (0..1)
    const int tt0   = (wave >> 1) * 32;          // 2 subtiles of 16

    v8f acc0 = {}, acc1 = {};
    #pragma unroll 4
    for (int kk = 0; kk < QC / 32; ++kk) {       // 32 steps, 64 WMMAs/wave
        int q0 = kk * 32;
        int row = kk >> 1;                       // channel
        int jbase = (kk & 1) * 32;
        v16h a  = load_a_frag<QC>(Ag, mtile * 16 + nl, q0, hf);
        const _Float16* bp = &sSig[row * RS + tt0 + nl + jbase + hf * 16];
        v16h b0 = load_b_frag(bp);
        v16h b1 = load_b_frag(bp + 16);
        acc0 = __builtin_amdgcn_wmma_f32_16x16x32_f16(
                   false, a, false, b0, (short)0, acc0, false, false);
        acc1 = __builtin_amdgcn_wmma_f32_16x16x32_f16(
                   false, a, false, b1, (short)0, acc1, false, false);
    }

    const float invL = 1.0f / Lptr[0];
    const float thr  = LMBD * invL;
    const int ta = t0 + tt0 + nl;

    #pragma unroll
    for (int i = 0; i < 8; ++i) {
        int ko = mtile * 16 + i + 8 * hf;
        size_t rowz = (size_t)(n * KA + ko);
        #pragma unroll
        for (int s = 0; s < 2; ++s) {
            int t = ta + s * 16;
            if (t < TZ) {
                float g = (s == 0) ? acc0[i] : acc1[i];
                size_t idx  = rowz * TZ + t;
                size_t pidx = rowz * TZP + 63 + t;
                float zo = z32[idx], wo = w32[idx];
                float wn = fmaxf(zo - g * invL - thr, 0.0f);
                float zn = wn + mom * (wn - wo);
                w32[idx] = wn;
                z32[idx] = zn;
                zpad[pidx] = (_Float16)zn;
                wpad[pidx] = (_Float16)wn;
            }
        }
    }
}

// ---------------- host orchestration ----------------
extern "C" void kernel_launch(void* const* d_in, const int* in_sizes, int n_in,
                              void* d_out, int out_size, void* d_ws, size_t ws_size,
                              hipStream_t stream) {
    (void)in_sizes; (void)n_in; (void)out_size; (void)ws_size;
    const float* x = (const float*)d_in[0];
    const float* D = (const float*)d_in[1];
    float* out = (float*)d_out;
    char* ws = (char*)d_ws;

    auto a256 = [](size_t v) { return (v + 255) & ~(size_t)255; };
    size_t off = 0;
    size_t oL  = off; off = a256(off + 4);
    size_t oDc = off; off = a256(off + (size_t)KA * QC * 2);
    size_t oDt = off; off = a256(off + (size_t)CH * QT * 2);
    size_t oZ  = off; off = a256(off + (size_t)NB * KA * TZ * 4);
    size_t oW  = off; off = a256(off + (size_t)NB * KA * TZ * 4);
    size_t oZP = off; off = a256(off + (size_t)NB * KA * TZP * 2);
    size_t oWP = off; off = a256(off + (size_t)NB * KA * TZP * 2);
    size_t oR  = off; off = a256(off + (size_t)NB * CH * T_LEN * 2);

    float*    Lp     = (float*)(ws + oL);
    _Float16* Dconv  = (_Float16*)(ws + oDc);
    _Float16* DconvT = (_Float16*)(ws + oDt);
    float*    z32    = (float*)(ws + oZ);
    float*    w32    = (float*)(ws + oW);
    _Float16* zpad   = (_Float16*)(ws + oZP);
    _Float16* wpad   = (_Float16*)(ws + oWP);
    _Float16* r16    = (_Float16*)(ws + oR);

    // zero FISTA state + padded f16 mirrors (word counts are even)
    k_zero<<<1024, 256, 0, stream>>>((unsigned int*)z32, (unsigned)(NB * KA * TZ));
    k_zero<<<1024, 256, 0, stream>>>((unsigned int*)w32, (unsigned)(NB * KA * TZ));
    k_zero<<<1024, 256, 0, stream>>>((unsigned int*)zpad, (unsigned)(NB * KA * TZP / 2));
    k_zero<<<1024, 256, 0, stream>>>((unsigned int*)wpad, (unsigned)(NB * KA * TZP / 2));

    k_prepD<<<(KA * CH * KS + 255) / 256, 256, 0, stream>>>(D, Dconv, DconvT);
    k_lipschitz<<<1, 256, 0, stream>>>(D, Lp);

    float beta = 1.0f;
    for (int it = 0; it < 20; ++it) {
        // r = convt(z, D) - x
        k_convt<0><<<dim3(T_LEN / 256, NB), 256, 0, stream>>>(
            zpad, DconvT, x, r16, nullptr);
        float bn = (1.0f + sqrtf(1.0f + 4.0f * beta * beta)) * 0.5f;
        float mom = (beta - 1.0f) / bn;
        beta = bn;
        // grad = conv(r, D); prox + momentum; refresh f16 mirrors
        k_conv_prox<<<dim3((TZ + 127) / 128, NB), 256, 0, stream>>>(
            r16, Dconv, Lp, z32, w32, zpad, wpad, mom);
    }
    // final: out = convt(w, D)
    k_convt<1><<<dim3(T_LEN / 256, NB), 256, 0, stream>>>(
        wpad, DconvT, x, nullptr, out);
}
